// MixtureLinear_30356828848765
// MI455X (gfx1250) — matrix operand order repaired and checked by reading, not compiled, not run on hardware
//
#include <hip/hip_runtime.h>

typedef _Float16 v4h  __attribute__((ext_vector_type(4)));
typedef _Float16 v8h  __attribute__((ext_vector_type(8)));
typedef _Float16 v16h __attribute__((ext_vector_type(16)));
typedef float    v8f  __attribute__((ext_vector_type(8)));
typedef unsigned int u32x4 __attribute__((ext_vector_type(4)));
typedef int      i32x4 __attribute__((ext_vector_type(4)));
typedef int      i32x8 __attribute__((ext_vector_type(8)));

// Problem sizes (fixed by the reference)
constexpr int Bb = 8, Nn = 1024, Cc = 1024, Dd = 1024, Rr = 8;
constexpr int MM = Bb * Nn;   // 8192  GEMM M  (rows m = b*N + n)
constexpr int KK = Cc * Rr;   // 8192  GEMM K  (k = r*1024 + c)
// GEMM N-dim = Dd = 1024

// GEMM tiling
constexpr int WGM = 128;      // block tile M
constexpr int WGN = 128;      // block tile N (2x2 waves, 64x64 per wave)
constexpr int KT  = 32;       // k-step (one f16 WMMA K)
constexpr int LDA = 40;       // LDS row stride (halves): 80B = 16B-aligned, conflict-free
constexpr int LDB = 40;
constexpr int NT  = KK / KT;  // 256 k-steps

#if __has_builtin(__builtin_amdgcn_tensor_load_to_lds)
#define USE_TDM 1
#endif

// ---------------- preprocessing kernels ----------------

// fp32 input -> f16, same [M, C] layout
__global__ __launch_bounds__(256)
void cvt_in(const float* __restrict__ in, _Float16* __restrict__ outh)
{
    const size_t i = ((size_t)blockIdx.x * 256 + threadIdx.x) * 4;
    const float4 v = *(const float4*)(in + i);
    v4h h = { (_Float16)v.x, (_Float16)v.y, (_Float16)v.z, (_Float16)v.w };
    *(v4h*)(outh + i) = h;
}

// weight[D][C][R] fp32 -> Bmat[d][k] f16 with k = r*1024 + c
__global__ __launch_bounds__(256)
void cvt_w(const float* __restrict__ w, _Float16* __restrict__ Bw)
{
    __shared__ _Float16 t[8][33];
    const int d   = blockIdx.y;
    const int c0  = blockIdx.x * 32;
    const int tid = threadIdx.x;
    const float x = w[(size_t)d * (Cc * Rr) + (size_t)c0 * Rr + tid]; // (c,r) r-fastest
    t[tid & 7][tid >> 3] = (_Float16)x;
    __syncthreads();
    const int r = tid >> 5, c = tid & 31;
    Bw[(size_t)d * KK + (size_t)r * Cc + (c0 + c)] = t[r][c];
}

// biasT[n][d] = sum_r coef[n,r] * bias[d,r]   (fp32, 4MB)
__global__ __launch_bounds__(256)
void bias_mix(const float* __restrict__ coef, const float* __restrict__ bias,
              float* __restrict__ biasT)
{
    const int i = blockIdx.x * 256 + threadIdx.x;   // over Nn*Dd
    const int n = i >> 10, d = i & (Dd - 1);
    float s = 0.f;
#pragma unroll
    for (int r = 0; r < Rr; ++r) s = fmaf(coef[n * Rr + r], bias[d * Rr + r], s);
    biasT[i] = s;
}

// ---------------- main fused GEMM ----------------
// out[m,d] = sum_k A'[m,k] * B'[k,d] + biasT[n(m),d]
// A'[m, r*1024+c] = Ah[m,c] * coef[n(m), r]   (scale applied at LDS-stage time)
// Double-buffered LDS, TDM prefetch of B tiles overlapped with WMMA.

__global__ __launch_bounds__(128)
void mixgemm(const _Float16* __restrict__ Ah, const _Float16* __restrict__ Bw,
             const float* __restrict__ coef, const float* __restrict__ biasT,
             float* __restrict__ out)
{
    __shared__ _Float16 As[2][WGM][LDA];   // 2 x 10 KB
    __shared__ _Float16 Bs[2][WGN][LDB];   // 2 x 10 KB

    const int tid   = threadIdx.x;
    const int lane  = tid & 31;
    const int wave  = tid >> 5;        // 4 waves: 2 (M) x 2 (N)
    const int wm    = wave >> 1;
    const int wn    = wave & 1;
    const int m0    = blockIdx.y * WGM;
    const int d0    = blockIdx.x * WGN;
    const int row16 = lane & 15;
    const int hsel  = lane >> 4;       // lane half-select per ISA WMMA layouts

    // this thread's A staging row (one row per thread)
    const int mrow = m0 + tid;
    const int nrow = mrow & (Nn - 1);
    const float*    coefrow = coef + nrow * Rr;
    const _Float16* arow    = Ah + (size_t)mrow * Cc;

    v8f acc[4][4] = {};                // each wave: 64(M) x 64(N) of output

    // ---- staging helpers ----
    auto issueB = [&](int kt2, int nbuf) {
#if USE_TDM
        if (tid < 32) {  // one wave issues the DMA (EXEC-independent op)
            const unsigned long long ga =
                (unsigned long long)(Bw + (size_t)d0 * KK + kt2 * KT);
            const unsigned ldsb = (unsigned)(unsigned long long)&Bs[nbuf][0][0];
            u32x4 g0 = {};
            g0[0] = 1u;                                    // count=1, user mode
            g0[1] = ldsb;                                  // lds_addr
            g0[2] = (unsigned)ga;                          // global_addr[31:0]
            g0[3] = ((unsigned)(ga >> 32) & 0x01FFFFFFu)   // global_addr[56:32]
                    | 0x80000000u;                         // type=2 ("image")
            i32x8 g1 = {};
            // data_size=1 (2B), pad_enable, pad_interval=16 DW (row=64B),
            // pad_amount=4 DW (8 halves -> LDB-KT padding)
            g1[0] = (1 << 16) | (1 << 20) | (3 << 22) | (3 << 25);
            g1[1] = (KK & 0xFFFF) << 16;                   // tensor_dim0 lo
            g1[2] = (KK >> 16) | ((Dd & 0xFFFF) << 16);    // dim0 hi | tensor_dim1 lo
            g1[3] = (Dd >> 16) | (KT << 16);               // dim1 hi | tile_dim0=32
            g1[4] = WGN;                                   // tile_dim1=128, tile_dim2=0
            g1[5] = KK;                                    // tensor_dim0_stride lo
            g1[6] = 0;
            g1[7] = 0;
            i32x4 g2 = {}, g3 = {};
#if __clang_major__ >= 23
            i32x8 g4 = {};
            __builtin_amdgcn_tensor_load_to_lds(g0, g1, g2, g3, g4, 0);
#else
            __builtin_amdgcn_tensor_load_to_lds(g0, g1, g2, g3, 0);
#endif
        }
#else
        {   // fallback: one thread per d-row, 64B each
            const v8h* src = (const v8h*)(Bw + (size_t)(d0 + tid) * KK + kt2 * KT);
            v8h b0 = src[0], b1 = src[1], b2 = src[2], b3 = src[3];
            v8h* dst = (v8h*)&Bs[nbuf][tid][0];
            dst[0] = b0; dst[1] = b1; dst[2] = b2; dst[3] = b3;
        }
#endif
    };

    v8h      ax[4];                    // staged A row (raw, pre-scale)
    _Float16 sc = (_Float16)0.f;       // coef scale for the staged slice

    auto loadA = [&](int kt2) {        // issue global loads for k-step kt2
        const int c0 = (kt2 * KT) & (Cc - 1);
        const v8h* src = (const v8h*)(arow + c0);
        ax[0] = src[0]; ax[1] = src[1]; ax[2] = src[2]; ax[3] = src[3];
        sc = (_Float16)coefrow[kt2 >> 5];                  // r = kt2*KT / 1024
        if (kt2 + 2 < NT)
            __builtin_prefetch(arow + (((kt2 + 2) * KT) & (Cc - 1)), 0, 1);
    };
    auto storeA = [&](int nbuf) {      // scale + store into LDS
        v8h* dst = (v8h*)&As[nbuf][tid][0];
        dst[0] = ax[0] * sc; dst[1] = ax[1] * sc;
        dst[2] = ax[2] * sc; dst[3] = ax[3] * sc;
    };

    // ---- prologue: fill buffer 0 ----
    issueB(0, 0);
    loadA(0);
    storeA(0);
#if USE_TDM
    if (tid < 32) {
#if __has_builtin(__builtin_amdgcn_s_wait_tensorcnt)
        __builtin_amdgcn_s_wait_tensorcnt(0);
#else
        asm volatile("s_wait_tensorcnt 0x0" ::: "memory");
#endif
    }
#endif
    __syncthreads();

    // ---- main pipelined loop: one barrier per k-step ----
    for (int kt = 0; kt < NT; ++kt) {
        const int buf = kt & 1;
        const bool more = (kt + 1) < NT;

        if (more) {
            issueB(kt + 1, buf ^ 1);   // async DMA into next buffer
            loadA(kt + 1);             // global loads in flight during WMMA
        }

        // ---- compute current buffer (contiguous k = 16*hsel + s in A and B fragments) ----
        v16h bfr[4];
#pragma unroll
        for (int tn = 0; tn < 4; ++tn) {
            // B 32x16: lanes 0-15 col=l K=0..15; lanes 16-31 col=l-16 K=16..31
            const _Float16* p = &Bs[buf][wn * 64 + tn * 16 + row16][hsel * 16];
            v8h lo = *(const v8h*)p;
            v8h hi = *(const v8h*)(p + 8);
            bfr[tn] = __builtin_shufflevector(lo, hi, 0,1,2,3,4,5,6,7,8,9,10,11,12,13,14,15);
        }
#pragma unroll
        for (int tm = 0; tm < 4; ++tm) {
            // A 16x32: lane<16 row=l holds K 0..15; lane>=16 holds K 16..31
            const _Float16* p = &As[buf][wm * 64 + tm * 16 + row16][hsel * 16];
            v8h lo = *(const v8h*)p;
            v8h hi = *(const v8h*)(p + 8);
            v16h af = __builtin_shufflevector(lo, hi, 0,1,2,3,4,5,6,7,8,9,10,11,12,13,14,15);
#pragma unroll
            for (int tn = 0; tn < 4; ++tn)
                acc[tm][tn] = __builtin_amdgcn_wmma_f32_16x16x32_f16(
                    false, af, false, bfr[tn], (short)0, acc[tm][tn], false, false);
        }

        if (more) storeA(buf ^ 1);     // waits loadcnt here, after WMMA issued

#if USE_TDM
        if (tid < 32) {
#if __has_builtin(__builtin_amdgcn_s_wait_tensorcnt)
            __builtin_amdgcn_s_wait_tensorcnt(0);
#else
            asm volatile("s_wait_tensorcnt 0x0" ::: "memory");
#endif
        }
#endif
        __syncthreads();               // reads of buf are DScnt-waited before signal
    }

    // ---- epilogue: + biasT, coalesced stores (16 consecutive d per lane group) ----
#pragma unroll
    for (int tm = 0; tm < 4; ++tm) {
#pragma unroll
        for (int tn = 0; tn < 4; ++tn) {
            const int d = d0 + wn * 64 + tn * 16 + row16;
            v8f a = acc[tm][tn];
#pragma unroll
            for (int v = 0; v < 8; ++v) {
                const int m = m0 + wm * 64 + tm * 16 + v + hsel * 8; // C/D: VGPR v -> M=v / v+8
                const int n = m & (Nn - 1);
                out[(size_t)m * Dd + d] = a[v] + biasT[(size_t)n * Dd + d];
            }
        }
    }
}

// ---------------- host-side launch ----------------

extern "C" void kernel_launch(void* const* d_in, const int* in_sizes, int n_in,
                              void* d_out, int out_size, void* d_ws, size_t ws_size,
                              hipStream_t stream)
{
    (void)in_sizes; (void)n_in; (void)out_size; (void)ws_size;
    const float* input  = (const float*)d_in[0];   // [B, N, C]
    const float* weight = (const float*)d_in[1];   // [D, C, R]
    const float* bias   = (const float*)d_in[2];   // [D, R]
    const float* coef   = (const float*)d_in[3];   // [N, R]
    float* out = (float*)d_out;                    // [B, N, D]

    // workspace layout: 16MB f16 A | 16MB f16 B' | 4MB f32 biasT  (36MB total)
    char* ws = (char*)d_ws;
    _Float16* Ah    = (_Float16*)ws;
    _Float16* Bw    = (_Float16*)(ws + (size_t)MM * Cc * 2);
    float*    biasT = (float*)(ws + (size_t)MM * Cc * 2 + (size_t)Dd * KK * 2);

    cvt_in  <<<(MM * Cc) / 1024, 256, 0, stream>>>(input, Ah);
    cvt_w   <<<dim3(Cc / 32, Dd), 256, 0, stream>>>(weight, Bw);
    bias_mix<<<(Nn * Dd) / 256, 256, 0, stream>>>(coef, bias, biasT);
    mixgemm <<<dim3(Dd / WGN, MM / WGM), 128, 0, stream>>>(Ah, Bw, coef, biasT, out);
}
